// GlobalLocalAttention_39565238730989
// MI455X (gfx1250) — matrix operand order repaired and checked by reading, not compiled, not run on hardware
//
#include <hip/hip_runtime.h>

// ---------------------------------------------------------------------------
// GlobalLocalAttention on MI455X (gfx1250, wave32, WMMA bf16 16x16x32)
//
// Pipeline (5 kernels, all on `stream`):
//  1. cvt_weights : qkv_w / proj_w  f32 -> bf16
//  2. gate_tr     : g = clip((x*aw+ab)*x,0,6), stored bf16 pixel-major [p][c]
//  3. qkv_gemm    : [768x256]x[256xP] bf16 WMMA, scatter-store q,k (tok-major)
//                   and v (dim-major = transposed) per (window, head)
//  4. attn_win    : per (window,head): Q@K^T (K padded 16->32), f32 softmax,
//                   P@V via LDS relayout; store bf16 pixel-major [p][c]
//  5. proj_gemm   : [256x256]x[256xP] bf16 WMMA + bias -> f32 NCHW output
//
// global_prefetch_b8 (via __builtin_prefetch) issued for streamed operand rows.
// ---------------------------------------------------------------------------

typedef __bf16 bf16;
typedef __attribute__((ext_vector_type(16))) __bf16 bf16x16;
typedef __attribute__((ext_vector_type(8)))  __bf16 bf16x8;
typedef __attribute__((ext_vector_type(8)))  float  f32x8;

#define WMMA_BF16(a, b, c) \
  __builtin_amdgcn_wmma_f32_16x16x32_bf16(false, (a), false, (b), (short)0, (c), false, false)

union AOp { bf16x16 v; bf16x8 h[2]; };

static __device__ __forceinline__ unsigned short f2bf(float f) {
  unsigned u = __builtin_bit_cast(unsigned, f);
  u += 0x7FFFu + ((u >> 16) & 1u);           // round-to-nearest-even
  return (unsigned short)(u >> 16);
}
static __device__ __forceinline__ bf16x16 ld16(const unsigned short* p) {
  return *reinterpret_cast<const bf16x16*>(p);
}
static __device__ __forceinline__ bf16x8 ld8(const unsigned short* p) {
  return *reinterpret_cast<const bf16x8*>(p);
}

// Sizes
#define HW   65536       // 256*256 pixels per image
#define NPIX 131072      // B*H*W
#define QS   33554432    // elements per q/k/v plane = B*C*H*W

// ---------------------------------------------------------------- weights ---
__global__ void cvt_weights(const float* __restrict__ qkv_w,
                            const float* __restrict__ proj_w,
                            unsigned short* __restrict__ w1,
                            unsigned short* __restrict__ w2) {
  const int n1 = 768 * 256, n2 = 256 * 256;
  for (int i = blockIdx.x * blockDim.x + threadIdx.x; i < n1 + n2;
       i += gridDim.x * blockDim.x) {
    if (i < n1) w1[i] = f2bf(qkv_w[i]);
    else        w2[i - n1] = f2bf(proj_w[i - n1]);
  }
}

// ------------------------------------------------- gate + transpose to bf16 --
// x[b][c][p] (coalesced read over p) -> g_t[b*HW+p][c] (coalesced write over c)
__global__ __launch_bounds__(256) void gate_tr(const float* __restrict__ x,
                                               const float* __restrict__ aw,
                                               const float* __restrict__ ab,
                                               unsigned short* __restrict__ Gt) {
  __shared__ unsigned short tile[32][33];
  const int b = blockIdx.z, ct = blockIdx.y * 32, ptb = blockIdx.x * 32;
  const int tx = threadIdx.x, ty = threadIdx.y;   // block (32, 8)
  for (int k = 0; k < 4; ++k) {
    const int cc = ct + ty + k * 8;
    const float xv = x[((b * 256 + cc) << 16) + ptb + tx];
    float g = (xv * aw[cc] + ab[cc]) * xv;
    g = fminf(fmaxf(g, 0.0f), 6.0f);
    tile[ty + k * 8][tx] = f2bf(g);
  }
  __syncthreads();
  for (int k = 0; k < 4; ++k) {
    const int pp = ptb + ty + k * 8;
    Gt[(((b << 16) + pp) << 8) + ct + tx] = tile[tx][ty + k * 8];
  }
}

// --------------------------------------------------------------- qkv GEMM ---
// Per wave: 16 output rows x 64 pixels, K=256 (8 WMMA k-steps).
// Scatter-store into windowed layout:
//   q,k : [win][head][tok(64)][d(16)]   v : [win][head][d(16)][tok(64)]
__global__ __launch_bounds__(256) void qkv_gemm(const unsigned short* __restrict__ Wq,
                                                const unsigned short* __restrict__ Gt,
                                                unsigned short* __restrict__ QKV) {
  const int lane = threadIdx.x & 31, wv = threadIdx.x >> 5;
  const bool hi = lane >= 16;
  const int l16 = lane & 15;
  const int rt = blockIdx.y * 8 + wv;   // 0..47  (768 rows / 16)
  const int pbase = blockIdx.x * 64;    // pixel tile
  const unsigned short* arow = Wq + (rt * 16 + l16) * 256;

  // Speculative prefetch of the streamed 512B operand rows (global_prefetch_b8;
  // does not touch LOADcnt, hides first-touch HBM latency).
  __builtin_prefetch(arow, 0, 3);
  __builtin_prefetch(arow + 128, 0, 3);
  for (int j = 0; j < 4; ++j) {
    const unsigned short* brow = Gt + (pbase + j * 16 + l16) * 256;
    __builtin_prefetch(brow, 0, 3);
    __builtin_prefetch(brow + 128, 0, 3);
  }

  f32x8 acc[4] = {{}, {}, {}, {}};
  for (int kk = 0; kk < 8; ++kk) {
    AOp a;
    const int ao = kk * 32 + (hi ? 16 : 0);
    a.h[0] = ld8(arow + ao);
    a.h[1] = ld8(arow + ao + 8);
    const int bo = kk * 32 + (hi ? 16 : 0);
    for (int j = 0; j < 4; ++j) {
      bf16x16 b = ld16(Gt + (pbase + j * 16 + l16) * 256 + bo);
      acc[j] = WMMA_BF16(a.v, b, acc[j]);
    }
  }
  for (int j = 0; j < 4; ++j) {
    const int P = pbase + j * 16 + l16;
    const int b = P >> 16, pix = P & 65535;
    const int h = pix >> 8, wc = pix & 255;
    const int win = (b << 10) + ((h >> 3) << 5) + (wc >> 3);
    const int tok = ((h & 7) << 3) + (wc & 7);
    for (int i = 0; i < 8; ++i) {
      const int dfull = rt * 16 + i + (hi ? 8 : 0);   // 0..767
      const int which = dfull >> 8;                   // 0=q 1=k 2=v
      const int n = (dfull & 255) >> 4, d = dfull & 15;
      int idx;
      if (which == 2) idx = ((win * 16 + n) * 16 + d) * 64 + tok;  // v transposed
      else            idx = ((win * 16 + n) * 64 + tok) * 16 + d;
      QKV[which * QS + idx] = f2bf(acc[j][i]);
    }
  }
}

// --------------------------------------------------------------- attention --
// One block per (window, head); 4 waves = 4 query tiles of 16 rows.
__global__ __launch_bounds__(128) void attn_win(const unsigned short* __restrict__ QKV,
                                                unsigned short* __restrict__ Ot) {
  const int win = blockIdx.x, head = blockIdx.y;
  const int lane = threadIdx.x & 31, mt = threadIdx.x >> 5;
  const bool hi = lane >= 16;
  const int l16 = lane & 15;
  __shared__ unsigned short Plds[64][80];   // 64x64 probs, +16 shorts row pad

  const unsigned short* q = QKV + (win * 16 + head) * 1024;            // [64][16]
  const unsigned short* k = QKV + QS + (win * 16 + head) * 1024;       // [64][16]
  const unsigned short* v = QKV + 2 * QS + (win * 16 + head) * 1024;   // [16][64]

  // Prefetch this head-window's K and V blocks (2KB each, 64B/lane).
  __builtin_prefetch(k + lane * 32, 0, 3);
  __builtin_prefetch(v + lane * 32, 0, 3);

  // ---- S = Q @ K^T  (hd=16 padded to K=32 with zeros) ----
  AOp a;
  a.h[0] = hi ? bf16x8{} : ld8(q + (mt * 16 + l16) * 16);
  a.h[1] = hi ? bf16x8{} : ld8(q + (mt * 16 + l16) * 16 + 8);   // hi lanes = K 16..31 pad
  f32x8 s[4] = {{}, {}, {}, {}};
  for (int nt = 0; nt < 4; ++nt) {
    bf16x16 bk = {};
    if (!hi) bk = ld16(k + (nt * 16 + l16) * 16);  // hi lanes = K 16..31 pad
    s[nt] = WMMA_BF16(a.v, bk, s[nt]);
  }

  // ---- softmax over 64 keys (scale = hd^-0.5 = 0.25 folded into exp2) ----
  const float cexp = 0.25f * 1.44269504f;
  for (int i = 0; i < 8; ++i) {
    float m = fmaxf(fmaxf(s[0][i], s[1][i]), fmaxf(s[2][i], s[3][i]));
    for (int off = 8; off; off >>= 1) m = fmaxf(m, __shfl_xor(m, off, 16));
    float e0 = exp2f((s[0][i] - m) * cexp);
    float e1 = exp2f((s[1][i] - m) * cexp);
    float e2 = exp2f((s[2][i] - m) * cexp);
    float e3 = exp2f((s[3][i] - m) * cexp);
    float sum = e0 + e1 + e2 + e3;
    for (int off = 8; off; off >>= 1) sum += __shfl_xor(sum, off, 16);
    const float r = 1.0f / sum;
    const int row = mt * 16 + i + (hi ? 8 : 0);
    Plds[row][l16]      = f2bf(e0 * r);
    Plds[row][16 + l16] = f2bf(e1 * r);
    Plds[row][32 + l16] = f2bf(e2 * r);
    Plds[row][48 + l16] = f2bf(e3 * r);
  }
  __syncthreads();

  // ---- O = P @ V  (K=64 tokens, 2 WMMA k-steps; V is dim-major) ----
  f32x8 o = {};
  const int prow = mt * 16 + l16;
  for (int kb = 0; kb < 64; kb += 32) {
    AOp pa;
    pa.h[0] = ld8(&Plds[prow][kb + (hi ? 16 : 0)]);
    pa.h[1] = ld8(&Plds[prow][kb + 8 + (hi ? 16 : 0)]);
    bf16x16 bv = ld16(v + l16 * 64 + kb + (hi ? 16 : 0));
    o = WMMA_BF16(pa.v, bv, o);
  }

  // ---- store pixel-major bf16 for the proj GEMM ----
  const int b = win >> 10, wl = win & 1023;
  const int wh = wl >> 5, ww = wl & 31;
  for (int i = 0; i < 8; ++i) {
    const int t = mt * 16 + i + (hi ? 8 : 0);
    const int hrow = wh * 8 + (t >> 3), wcol = ww * 8 + (t & 7);
    const int p = (hrow << 8) + wcol;
    Ot[(((b << 16) + p) << 8) + head * 16 + l16] = f2bf(o[i]);
  }
}

// -------------------------------------------------------------- proj GEMM ---
__global__ __launch_bounds__(256) void proj_gemm(const unsigned short* __restrict__ Wp,
                                                 const unsigned short* __restrict__ Ot,
                                                 const float* __restrict__ bias,
                                                 float* __restrict__ out) {
  const int lane = threadIdx.x & 31, wv = threadIdx.x >> 5;
  const bool hi = lane >= 16;
  const int l16 = lane & 15;
  const int rt = blockIdx.y * 8 + wv;   // 0..15 (256 rows / 16)
  const int pbase = blockIdx.x * 64;
  const unsigned short* arow = Wp + (rt * 16 + l16) * 256;

  __builtin_prefetch(arow, 0, 3);
  __builtin_prefetch(arow + 128, 0, 3);
  for (int j = 0; j < 4; ++j) {
    const unsigned short* brow = Ot + (pbase + j * 16 + l16) * 256;
    __builtin_prefetch(brow, 0, 3);
    __builtin_prefetch(brow + 128, 0, 3);
  }

  f32x8 acc[4] = {{}, {}, {}, {}};
  for (int kk = 0; kk < 8; ++kk) {
    AOp a;
    const int ao = kk * 32 + (hi ? 16 : 0);
    a.h[0] = ld8(arow + ao);
    a.h[1] = ld8(arow + ao + 8);
    const int bo = kk * 32 + (hi ? 16 : 0);
    for (int j = 0; j < 4; ++j) {
      bf16x16 b = ld16(Ot + (pbase + j * 16 + l16) * 256 + bo);
      acc[j] = WMMA_BF16(a.v, b, acc[j]);
    }
  }
  for (int j = 0; j < 4; ++j) {
    const int P = pbase + j * 16 + l16;
    const int b = P >> 16, pix = P & 65535;
    for (int i = 0; i < 8; ++i) {
      const int dout = rt * 16 + i + (hi ? 8 : 0);
      out[b * 16777216 + (dout << 16) + pix] = acc[j][i] + bias[dout];
    }
  }
}

// ------------------------------------------------------------------ launch ---
extern "C" void kernel_launch(void* const* d_in, const int* in_sizes, int n_in,
                              void* d_out, int out_size, void* d_ws, size_t ws_size,
                              hipStream_t stream) {
  const float* x      = (const float*)d_in[0];
  const float* ac_w   = (const float*)d_in[1];
  const float* ac_b   = (const float*)d_in[2];
  const float* qkv_w  = (const float*)d_in[3];
  const float* proj_w = (const float*)d_in[4];
  const float* proj_b = (const float*)d_in[5];
  float* out = (float*)d_out;

  char* ws = (char*)d_ws;
  unsigned short* w1  = (unsigned short*)(ws);                               // 768*256 bf16
  unsigned short* w2  = (unsigned short*)(ws + 393216);                      // 256*256 bf16
  unsigned short* Gt  = (unsigned short*)(ws + 524288);                      // 131072*256 bf16
  unsigned short* QKV = (unsigned short*)(ws + 524288 + 67108864);           // 3*QS bf16
  unsigned short* Ot  = (unsigned short*)(ws + 524288 + 67108864 + 201326592);

  cvt_weights<<<256, 256, 0, stream>>>(qkv_w, proj_w, w1, w2);
  gate_tr<<<dim3(2048, 8, 2), dim3(32, 8), 0, stream>>>(x, ac_w, ac_b, Gt);
  qkv_gemm<<<dim3(2048, 6), 256, 0, stream>>>(w1, Gt, QKV);
  attn_win<<<dim3(2048, 16), 128, 0, stream>>>(QKV, Ot);
  proj_gemm<<<dim3(2048, 2), 256, 0, stream>>>(w2, Ot, proj_b, out);
}
